// MultiHeadSparseAttentionSBH_25769803776758
// MI455X (gfx1250) — matrix operand order repaired and checked by reading, not compiled, not run on hardware
//
#include <hip/hip_runtime.h>
#include <hip/hip_bf16.h>

// ---------------------------------------------------------------------------
// MI455X (gfx1250) sparse multi-head attention, SBH layout.
//   S=8192, B=2, H=16, Dh=96, D=1536, sparse n=4 -> SDPA with
//   batch'=8, heads=16, seq=G=2048, head_dim=96.
// Memory-bound (~0.6 GB min traffic @ 23.3 TB/s ~ 26 us; 0.52 TFLOP is noise
// for the WMMA array) -> bf16 WMMA + bf16 intermediates + flash attention.
// CDNA5 paths: v_wmma_f32_16x16x32_bf16 everywhere, async global->LDS copies
// (ASYNCcnt) double-buffering the K stream of the flash kernel.
// ---------------------------------------------------------------------------

typedef __attribute__((ext_vector_type(16))) __bf16 v16bf;
typedef __attribute__((ext_vector_type(8)))  __bf16 bf16x8;
typedef __attribute__((ext_vector_type(4)))  __bf16 bf16x4;
typedef __attribute__((ext_vector_type(8)))  float  v8f;
typedef __attribute__((ext_vector_type(4)))  float  f32x4;

#define S_TOT   8192
#define BATCH   2
#define NHEADS  16
#define HDIM    96
#define DMODEL  1536
#define MROWS   16384      // S*B
#define GLEN    2048       // S / n
#define NSLICE  128        // (n*B) * NHEADS
#define KTILES  48         // DMODEL / 32

// ===========================================================================
// Kernel 1: fused QKV projection (X @ W) + 3D RoPE (q,k) + scatter to
// attention layout [slice=j*16+h][g][d] in bf16.
// 64x64 output tile, 4 waves, wave tile 32x32 (2x2 of 16x16 WMMA), BK=32.
// ===========================================================================
__global__ __launch_bounds__(128)
void qkv_rope_kernel(const float* __restrict__ X,
                     const float* __restrict__ Wq,
                     const float* __restrict__ Wk,
                     const float* __restrict__ Wv,
                     __bf16* __restrict__ Qa,
                     __bf16* __restrict__ Ka,
                     __bf16* __restrict__ Va)
{
    __shared__ __align__(16) __bf16 As[64][40];   // [m][k], padded pitch
    __shared__ __align__(16) __bf16 Bs[64][40];   // [n][k] (transposed tile)

    const int t    = threadIdx.x;
    const int l    = t & 31;
    const int lr   = l & 15;
    const int lh   = l >> 4;
    const int wave = t >> 5;
    const int wm   = wave >> 1;
    const int wn   = wave & 1;

    const int mt = blockIdx.x;              // 0..255
    const int nt = blockIdx.y;              // 0..23
    const int z  = blockIdx.z;              // 0:q 1:k 2:v
    const float* W  = (z == 0) ? Wq : (z == 1) ? Wk : Wv;
    __bf16*     Dst = (z == 0) ? Qa : (z == 1) ? Ka : Va;

    const int rowBase = mt * 64;
    const int colBase = nt * 64;

    v8f acc[2][2] = {};

    for (int kt = 0; kt < KTILES; ++kt) {
        const int k0 = kt * 32;
        // ---- stage A tile 64x32 (f32 -> bf16) ----
#pragma unroll
        for (int i = 0; i < 4; ++i) {
            int idx = t + i * 128;          // 0..511 over 64x8 float4s
            int r   = idx >> 3;
            int kq  = (idx & 7) * 4;
            f32x4 v = *(const f32x4*)&X[(size_t)(rowBase + r) * DMODEL + k0 + kq];
            bf16x4 hv;
            hv[0] = (__bf16)v[0]; hv[1] = (__bf16)v[1];
            hv[2] = (__bf16)v[2]; hv[3] = (__bf16)v[3];
            *(bf16x4*)&As[r][kq] = hv;
        }
        // ---- stage B tile 32x64 transposed to [n][k] (f32 -> bf16) ----
#pragma unroll
        for (int i = 0; i < 4; ++i) {
            int idx = t + i * 128;          // 0..511 over 32x16 float4s
            int kk  = idx >> 4;
            int nq  = (idx & 15) * 4;
            f32x4 v = *(const f32x4*)&W[(size_t)(k0 + kk) * DMODEL + colBase + nq];
            Bs[nq + 0][kk] = (__bf16)v[0];
            Bs[nq + 1][kk] = (__bf16)v[1];
            Bs[nq + 2][kk] = (__bf16)v[2];
            Bs[nq + 3][kk] = (__bf16)v[3];
        }
        __syncthreads();

        // ---- fragments + WMMA ----
        v16bf a[2], b[2];
#pragma unroll
        for (int mi = 0; mi < 2; ++mi) {
            int m   = wm * 32 + mi * 16 + lr;
            int kk0 = lh * 8;
            bf16x8 lo = *(const bf16x8*)&As[m][kk0];
            bf16x8 hi = *(const bf16x8*)&As[m][kk0 + 16];
#pragma unroll
            for (int j = 0; j < 8; ++j) { a[mi][j] = lo[j]; a[mi][j + 8] = hi[j]; }
        }
#pragma unroll
        for (int ni = 0; ni < 2; ++ni) {
            int n   = wn * 32 + ni * 16 + lr;
            int kk0 = lh * 8;
            bf16x8 lo = *(const bf16x8*)&Bs[n][kk0];
            bf16x8 hi = *(const bf16x8*)&Bs[n][kk0 + 16];
#pragma unroll
            for (int j = 0; j < 8; ++j) { b[ni][j] = lo[j]; b[ni][j + 8] = hi[j]; }
        }
#pragma unroll
        for (int mi = 0; mi < 2; ++mi)
#pragma unroll
            for (int ni = 0; ni < 2; ++ni)
                acc[mi][ni] = __builtin_amdgcn_wmma_f32_16x16x32_bf16(
                    false, a[mi], false, b[ni], (short)0, acc[mi][ni], false, false);
        __syncthreads();
    }

    // ---- epilogue: RoPE (z<2) + scatter to attention layout ----
    // 32-wide, 32-aligned column tiles == one RoPE chunk (96 = 3*32); the
    // (d, d+16) rotation pair is (left frag, right frag) of the same lane.
    const int cbase = colBase + wn * 32;
    const int head  = cbase / HDIM;
    const int chunk = (cbase >> 5) % 3;     // 0:t 1:h 2:w
    const int dL    = chunk * 32 + lr;
    // 10000^(-lr/16) == exp2(-lr * log2(10000)/16); single v_exp_f32.
    const float invf = __builtin_amdgcn_exp2f((float)lr * -0.8304820237218407f);
#pragma unroll
    for (int mi = 0; mi < 2; ++mi) {
#pragma unroll
        for (int r = 0; r < 8; ++r) {
            int row  = rowBase + wm * 32 + mi * 16 + r + 8 * lh;
            int s    = row >> 1;
            int bix  = row & 1;
            float L  = acc[mi][0][r];
            float R  = acc[mi][1][r];
            float nL = L, nR = R;
            if (z < 2) {
                int pos = (chunk == 0) ? (s >> 10)
                        : (chunk == 1) ? ((s >> 5) & 31)
                                       : (s & 31);
                float ang = (float)pos * invf;
                float c, sn;
                __sincosf(ang, &sn, &c);
                nL = L * c - R * sn;
                nR = R * c + L * sn;
            }
            int g = s >> 2;
            int j = ((s & 3) << 1) | bix;   // token row = g*8 + j
            size_t base = ((size_t)(j * NHEADS + head) * GLEN + g) * HDIM;
            Dst[base + dL]      = (__bf16)nL;
            Dst[base + dL + 16] = (__bf16)nR;
        }
    }
}

// ===========================================================================
// Kernel 2: flash attention per (slice, 64-query tile). 4 waves x 16 queries.
// Q A-fragments cached in registers (loop-invariant). K tiles streamed with
// GLOBAL_LOAD_ASYNC_TO_LDS_B128 (ASYNCcnt), K/V double-buffered in LDS, next
// V tile staged in registers during compute. One barrier per KV step.
// LDS layout (single block => LDS offset 0 for the async VDST operand):
//   bytes [    0,13312) K buf0   [13312,26624) K buf1     (pitch 208 B)
//   bytes [26624,40448) Vt buf0  [40448,54272) Vt buf1    (pitch 144 B)
//   bytes [54272,63488) P (per-wave 16x72 bf16)           (pitch 144 B)
// ===========================================================================
#define KS_B(buf)   ((buf) * 13312)
#define VT_B(buf)   (26624 + (buf) * 13824)
#define PS_B        54272
#define KS_E(buf)   ((buf) * 6656)          // element offsets (bf16 units)
#define VT_E(buf)   (13312 + (buf) * 6912)
#define PS_E        27136

__global__ __launch_bounds__(128)
void flash_attn_kernel(const __bf16* __restrict__ Qa,
                       const __bf16* __restrict__ Ka,
                       const __bf16* __restrict__ Va,
                       __bf16* __restrict__ O)
{
    __shared__ __align__(16) __bf16 smem[31744];    // 63488 bytes

    const int t  = threadIdx.x;
    const int l  = t & 31;
    const int lr = l & 15;
    const int lh = l >> 4;
    const int wq = t >> 5;                  // wave -> 16-query strip
    const int gt = blockIdx.x;              // 0..31 query tile
    const int slice = blockIdx.y;           // 0..127 = j*16 + h

    const size_t sbase = (size_t)slice * GLEN * HDIM;
    const unsigned long long kbase = (unsigned long long)(Ka + sbase);

    // per-thread async copy chunk coordinates (64 rows x 12 x bf16x8)
    int cr[6], cd[6];
#pragma unroll
    for (int i = 0; i < 6; ++i) {
        int idx = t + i * 128;
        cr[i] = idx / 12;
        cd[i] = (idx % 12) * 8;
    }

    // ---- Q A-fragments: loop-invariant, straight from global to registers --
    v16bf qa[3];
    {
        const __bf16* qrow = Qa + sbase + (size_t)(gt * 64 + wq * 16 + lr) * HDIM;
#pragma unroll
        for (int kf = 0; kf < 3; ++kf) {
            int kk0 = kf * 32 + lh * 8;
            bf16x8 lo = *(const bf16x8*)&qrow[kk0];
            bf16x8 hi = *(const bf16x8*)&qrow[kk0 + 16];
#pragma unroll
            for (int j = 0; j < 8; ++j) { qa[kf][j] = lo[j]; qa[kf][j + 8] = hi[j]; }
        }
    }

    // ---- async K tile copy: global (bf16, row-major) -> LDS, 16B per lane --
    auto issueK = [&](int kt, int buf) {
#pragma unroll
        for (int i = 0; i < 6; ++i) {
            unsigned ldsOff = (unsigned)(KS_B(buf) + cr[i] * 208 + cd[i] * 2);
            unsigned gOff   = (unsigned)(((kt * 64 + cr[i]) * HDIM + cd[i]) * 2);
            asm volatile("global_load_async_to_lds_b128 %0, %1, %2"
                         :: "v"(ldsOff), "v"(gOff), "s"(kbase)
                         : "memory");
        }
    };

    // ---- prologue: K0 async, V0 via registers (transposed store) ----
    issueK(0, 0);
#pragma unroll
    for (int i = 0; i < 6; ++i) {
        bf16x8 vv = *(const bf16x8*)&Va[sbase + (size_t)cr[i] * HDIM + cd[i]];
#pragma unroll
        for (int jj = 0; jj < 8; ++jj)
            smem[VT_E(0) + (cd[i] + jj) * 72 + cr[i]] = vv[jj];
    }
    asm volatile("s_wait_asynccnt 0x0" ::: "memory");
    __syncthreads();

    v8f acc[6] = {};
    float m_run = -3.0e38f;
    float l_run = 0.0f;
    const float scale = 0.10206207261596575f;   // 1/sqrt(96)

    for (int kt = 0; kt < 32; ++kt) {
        const int cur = kt & 1;
        const int nxt = cur ^ 1;
        const bool pre = (kt + 1 < 32);

        // ---- start next tile: V global loads into regs, K async into LDS --
        bf16x8 vstage[6];
        if (pre) {
#pragma unroll
            for (int i = 0; i < 6; ++i)
                vstage[i] = *(const bf16x8*)
                    &Va[sbase + (size_t)((kt + 1) * 64 + cr[i]) * HDIM + cd[i]];
            issueK(kt + 1, nxt);
        }

        // ---- S = Q K^T  (16 x 64 per wave) ----
        v8f sacc[4] = {};
#pragma unroll
        for (int kf = 0; kf < 3; ++kf) {
            int kd = kf * 32;
#pragma unroll
            for (int nf = 0; nf < 4; ++nf) {
                v16bf b;
                int n   = nf * 16 + lr;
                int kk0 = kd + lh * 8;
                bf16x8 lo = *(const bf16x8*)&smem[KS_E(cur) + n * 104 + kk0];
                bf16x8 hi = *(const bf16x8*)&smem[KS_E(cur) + n * 104 + kk0 + 16];
#pragma unroll
                for (int j = 0; j < 8; ++j) { b[j] = lo[j]; b[j + 8] = hi[j]; }
                sacc[nf] = __builtin_amdgcn_wmma_f32_16x16x32_bf16(
                    false, qa[kf], false, b, (short)0, sacc[nf], false, false);
            }
        }

        // ---- spill S to LDS (bf16), online softmax in place ----
#pragma unroll
        for (int nf = 0; nf < 4; ++nf)
#pragma unroll
            for (int r = 0; r < 8; ++r)
                smem[PS_E + (wq * 16 + r + 8 * lh) * 72 + nf * 16 + lr] =
                    (__bf16)sacc[nf][r];

        const int srow = lr;                // lane owns row lr, cols lh*32..+31
        const int c0   = lh * 32;
        float mloc = -3.0e38f;
#pragma unroll
        for (int c = 0; c < 32; ++c) {
            float v = (float)smem[PS_E + (wq * 16 + srow) * 72 + c0 + c] * scale;
            mloc = fmaxf(mloc, v);
        }
        mloc = fmaxf(mloc, __shfl_xor(mloc, 16, 32));
        float mnew  = fmaxf(m_run, mloc);
        float alpha = __expf(m_run - mnew);
        float lsum  = 0.0f;
#pragma unroll
        for (int c = 0; c < 32; ++c) {
            float v = (float)smem[PS_E + (wq * 16 + srow) * 72 + c0 + c] * scale;
            float p = __expf(v - mnew);
            lsum += p;
            smem[PS_E + (wq * 16 + srow) * 72 + c0 + c] = (__bf16)p;  // S -> P
        }
        lsum += __shfl_xor(lsum, 16, 32);
        l_run = l_run * alpha + lsum;
        m_run = mnew;

        // ---- rescale O accumulators by per-row alpha ----
#pragma unroll
        for (int r = 0; r < 8; ++r) {
            float a8 = __shfl(alpha, r + 8 * lh, 32);
#pragma unroll
            for (int f = 0; f < 6; ++f) acc[f][r] *= a8;
        }

        // ---- O += P @ V  (16 x 96 per wave) ----
#pragma unroll
        for (int ks = 0; ks < 64; ks += 32) {
            v16bf a;
            {
                int kk0 = ks + lh * 8;
                bf16x8 lo = *(const bf16x8*)&smem[PS_E + (wq * 16 + lr) * 72 + kk0];
                bf16x8 hi = *(const bf16x8*)&smem[PS_E + (wq * 16 + lr) * 72 + kk0 + 16];
#pragma unroll
                for (int j = 0; j < 8; ++j) { a[j] = lo[j]; a[j + 8] = hi[j]; }
            }
#pragma unroll
            for (int f = 0; f < 6; ++f) {
                v16bf b;
                int n   = f * 16 + lr;      // n = output dim d
                int kk0 = ks + lh * 8;
                bf16x8 lo = *(const bf16x8*)&smem[VT_E(cur) + n * 72 + kk0];
                bf16x8 hi = *(const bf16x8*)&smem[VT_E(cur) + n * 72 + kk0 + 16];
#pragma unroll
                for (int j = 0; j < 8; ++j) { b[j] = lo[j]; b[j + 8] = hi[j]; }
                acc[f] = __builtin_amdgcn_wmma_f32_16x16x32_bf16(
                    false, a, false, b, (short)0, acc[f], false, false);
            }
        }

        // ---- finish next tile: transposed V store, wait async K, barrier --
        if (pre) {
#pragma unroll
            for (int i = 0; i < 6; ++i)
#pragma unroll
                for (int jj = 0; jj < 8; ++jj)
                    smem[VT_E(nxt) + (cd[i] + jj) * 72 + cr[i]] = vstage[i][jj];
        }
        asm volatile("s_wait_asynccnt 0x0" ::: "memory");
        __syncthreads();
    }

    // ---- finalize: divide by l, scatter bf16 O in [token_row][D] layout ----
    float linv = 1.0f / l_run;
    const int jj = slice >> 4;
    const int hh = slice & 15;
#pragma unroll
    for (int r = 0; r < 8; ++r) {
        float iv = __shfl(linv, r + 8 * lh, 32);
        int m = r + 8 * lh;
        int g = gt * 64 + wq * 16 + m;
        size_t row = (size_t)g * 8 + jj;    // (g*n + k_off)*B + b == g*8 + j
#pragma unroll
        for (int f = 0; f < 6; ++f)
            O[row * DMODEL + hh * HDIM + f * 16 + lr] = (__bf16)(acc[f][r] * iv);
    }
}

// ===========================================================================
// Kernel 3: output projection  Y = O @ Wout + b_out  (bf16 x bf16 -> f32)
// ===========================================================================
__global__ __launch_bounds__(128)
void outproj_kernel(const __bf16* __restrict__ Om,
                    const float* __restrict__ Wout,
                    const float* __restrict__ bias,
                    float* __restrict__ Y)
{
    __shared__ __align__(16) __bf16 As[64][40];
    __shared__ __align__(16) __bf16 Bs[64][40];

    const int t    = threadIdx.x;
    const int l    = t & 31;
    const int lr   = l & 15;
    const int lh   = l >> 4;
    const int wave = t >> 5;
    const int wm   = wave >> 1;
    const int wn   = wave & 1;

    const int rowBase = blockIdx.x * 64;
    const int colBase = blockIdx.y * 64;

    v8f acc[2][2] = {};

    for (int kt = 0; kt < KTILES; ++kt) {
        const int k0 = kt * 32;
#pragma unroll
        for (int i = 0; i < 2; ++i) {       // A: already bf16
            int idx = t + i * 128;          // 0..255 over 64x4 bf16x8
            int r   = idx >> 2;
            int kq  = (idx & 3) * 8;
            bf16x8 v = *(const bf16x8*)&Om[(size_t)(rowBase + r) * DMODEL + k0 + kq];
            *(bf16x8*)&As[r][kq] = v;
        }
#pragma unroll
        for (int i = 0; i < 4; ++i) {       // B: f32 -> bf16, transpose
            int idx = t + i * 128;
            int kk  = idx >> 4;
            int nq  = (idx & 15) * 4;
            f32x4 v = *(const f32x4*)&Wout[(size_t)(k0 + kk) * DMODEL + colBase + nq];
            Bs[nq + 0][kk] = (__bf16)v[0];
            Bs[nq + 1][kk] = (__bf16)v[1];
            Bs[nq + 2][kk] = (__bf16)v[2];
            Bs[nq + 3][kk] = (__bf16)v[3];
        }
        __syncthreads();

        v16bf a[2], b[2];
#pragma unroll
        for (int mi = 0; mi < 2; ++mi) {
            int m   = wm * 32 + mi * 16 + lr;
            int kk0 = lh * 8;
            bf16x8 lo = *(const bf16x8*)&As[m][kk0];
            bf16x8 hi = *(const bf16x8*)&As[m][kk0 + 16];
#pragma unroll
            for (int j = 0; j < 8; ++j) { a[mi][j] = lo[j]; a[mi][j + 8] = hi[j]; }
        }
#pragma unroll
        for (int ni = 0; ni < 2; ++ni) {
            int n   = wn * 32 + ni * 16 + lr;
            int kk0 = lh * 8;
            bf16x8 lo = *(const bf16x8*)&Bs[n][kk0];
            bf16x8 hi = *(const bf16x8*)&Bs[n][kk0 + 16];
#pragma unroll
            for (int j = 0; j < 8; ++j) { b[ni][j] = lo[j]; b[ni][j + 8] = hi[j]; }
        }
#pragma unroll
        for (int mi = 0; mi < 2; ++mi)
#pragma unroll
            for (int ni = 0; ni < 2; ++ni)
                acc[mi][ni] = __builtin_amdgcn_wmma_f32_16x16x32_bf16(
                    false, a[mi], false, b[ni], (short)0, acc[mi][ni], false, false);
        __syncthreads();
    }

#pragma unroll
    for (int ni = 0; ni < 2; ++ni) {
        int col = colBase + wn * 32 + ni * 16 + lr;
        float bv = bias[col];
#pragma unroll
        for (int mi = 0; mi < 2; ++mi)
#pragma unroll
            for (int r = 0; r < 8; ++r) {
                int row = rowBase + wm * 32 + mi * 16 + r + 8 * lh;
                Y[(size_t)row * DMODEL + col] = acc[mi][ni][r] + bv;
            }
    }
}

// ===========================================================================
extern "C" void kernel_launch(void* const* d_in, const int* in_sizes, int n_in,
                              void* d_out, int out_size, void* d_ws, size_t ws_size,
                              hipStream_t stream)
{
    const float* X    = (const float*)d_in[0];
    const float* Wq   = (const float*)d_in[1];
    const float* Wk   = (const float*)d_in[2];
    const float* Wv   = (const float*)d_in[3];
    const float* Wout = (const float*)d_in[4];
    const float* bout = (const float*)d_in[5];

    const size_t attnElems = (size_t)NSLICE * GLEN * HDIM;   // 25,165,824
    __bf16* Qa = (__bf16*)d_ws;
    __bf16* Ka = Qa + attnElems;
    __bf16* Va = Ka + attnElems;
    __bf16* Om = Va + attnElems;                             // [MROWS][DMODEL] bf16

    dim3 g1(MROWS / 64, DMODEL / 64, 3);        // 256 x 24 x 3
    qkv_rope_kernel<<<g1, 128, 0, stream>>>(X, Wq, Wk, Wv, Qa, Ka, Va);

    dim3 g2(GLEN / 64, NSLICE);                 // 32 x 128
    flash_attn_kernel<<<g2, 128, 0, stream>>>(Qa, Ka, Va, Om);

    dim3 g3(MROWS / 64, DMODEL / 64);           // 256 x 24
    outproj_kernel<<<g3, 128, 0, stream>>>(Om, Wout, bout, (float*)d_out);
}
